// FEATHIST_56908316672538
// MI455X (gfx1250) — matrix operand order repaired and checked by reading, not compiled, not run on hardware
//
#include <hip/hip_runtime.h>
#include <math.h>

// HIST model on MI455X (gfx1250, wave32, WMMA).
// GEMMs: v_wmma_f32_16x16x32_f16, 32x64 tile/wave (8 WMMA per K-step),
// b128 vector loads on unit-stride A operands, global_prefetch on K-panels.
// Workspace requirement: ~93 MB.

#define NN 4096
#define CC 512
#define HH 128
#define NEGBIG (-3.402823466e38f)

typedef __attribute__((ext_vector_type(16))) _Float16 v16h;
typedef __attribute__((ext_vector_type(8)))  float    v8f;

// ---------------------------------------------------------------------------
// Fragment loaders.
// A 16x32 f16 fragment, lane (grp = lane>>4, rc = lane&15) covers row rc and
// K = kbase..kbase+7 (halves 0..7) plus kbase+16..kbase+23 (halves 8..15),
// kbase = kk + 8*grp.
// ---------------------------------------------------------------------------
__device__ __forceinline__ v16h fh_load_a_vec(const float* __restrict__ Arow, int kbase)
{
    const float4* p0 = (const float4*)(Arow + kbase);
    const float4* p1 = (const float4*)(Arow + kbase + 16);
    float4 f0 = p0[0], f1 = p0[1], f2 = p1[0], f3 = p1[1];
    v16h a;
    a[0]  = (_Float16)f0.x; a[1]  = (_Float16)f0.y; a[2]  = (_Float16)f0.z; a[3]  = (_Float16)f0.w;
    a[4]  = (_Float16)f1.x; a[5]  = (_Float16)f1.y; a[6]  = (_Float16)f1.z; a[7]  = (_Float16)f1.w;
    a[8]  = (_Float16)f2.x; a[9]  = (_Float16)f2.y; a[10] = (_Float16)f2.z; a[11] = (_Float16)f2.w;
    a[12] = (_Float16)f3.x; a[13] = (_Float16)f3.y; a[14] = (_Float16)f3.z; a[15] = (_Float16)f3.w;
    return a;
}

__device__ __forceinline__ v16h fh_load_a_str(const float* __restrict__ Arow, int kbase, long sak)
{
    v16h a;
#pragma unroll
    for (int h = 0; h < 16; ++h) {
        int k = kbase + ((h & 8) ? (h + 8) : h);
        a[h] = (_Float16)Arow[(long)k * sak];
    }
    return a;
}

// B 32x16 f16 fragment: lane holds N column rc, vec: halves are 16 contiguous K; str: same K order as A.
__device__ __forceinline__ v16h fh_load_b_vec(const float* __restrict__ Bp)
{
    const float4* p = (const float4*)Bp;
    float4 f0 = p[0], f1 = p[1], f2 = p[2], f3 = p[3];
    v16h b;
    b[0]  = (_Float16)f0.x; b[1]  = (_Float16)f0.y; b[2]  = (_Float16)f0.z; b[3]  = (_Float16)f0.w;
    b[4]  = (_Float16)f1.x; b[5]  = (_Float16)f1.y; b[6]  = (_Float16)f1.z; b[7]  = (_Float16)f1.w;
    b[8]  = (_Float16)f2.x; b[9]  = (_Float16)f2.y; b[10] = (_Float16)f2.z; b[11] = (_Float16)f2.w;
    b[12] = (_Float16)f3.x; b[13] = (_Float16)f3.y; b[14] = (_Float16)f3.z; b[15] = (_Float16)f3.w;
    return b;
}

__device__ __forceinline__ v16h fh_load_b_str(const float* __restrict__ Bp, long sbn)
{
    v16h b;
#pragma unroll
    for (int h = 0; h < 16; ++h)
        b[h] = (_Float16)Bp[(long)((h & 8) ? (h + 8) : h) * sbn];
    return b;
}

// ---------------------------------------------------------------------------
// Generic strided WMMA GEMM: D[m][n] = act( sum_k A(m,k)*B(k,n) + bias[n] ) (+D)
// A(m,k) = A[m*sam + k*sak], B(k,n) = B[k*sbk + n*sbn].
// One wave per 32x64 output tile: 2 A-frags x 4 B-frags = 8 WMMA per K-step.
// AK1: sak==1 (vector A loads); BN1: unused, B is always loaded strided.
// ---------------------------------------------------------------------------
template<bool AK1, bool BN1>
__global__ __launch_bounds__(32)
void fh_gemm_wmma(const float* __restrict__ A, long sam, long sak,
                  const float* __restrict__ B, long sbk, long sbn,
                  const float* __restrict__ bias,
                  float* __restrict__ D, long ldd,
                  int M, int Ncol, int Ktot, int act, int accum)
{
    const int lane = threadIdx.x & 31;
    const int grp  = lane >> 4;
    const int rc   = lane & 15;
    const long m0 = (long)blockIdx.y * 32;
    const long n0 = (long)blockIdx.x * 64;

    v8f acc[2][4] = {};
    const float* Arow0 = A + (m0 + rc) * sam;
    const float* Arow1 = A + (m0 + 16 + rc) * sam;

    for (int kk = 0; kk < Ktot; kk += 32) {
        const int kbase = kk + grp * 8;
        v16h a0 = AK1 ? fh_load_a_vec(Arow0, kbase) : fh_load_a_str(Arow0, kbase, sak);
        v16h a1 = AK1 ? fh_load_a_vec(Arow1, kbase) : fh_load_a_str(Arow1, kbase, sak);

        const float* Bk = B + (long)kbase * sbk;

        // speculative prefetch of next K-panel (global_prefetch_b8)
        if (kk + 32 < Ktot) {
            __builtin_prefetch(Arow0 + (long)(kk + 32 + grp * 8) * sak, 0, 1);
            __builtin_prefetch(Bk + (long)32 * sbk + n0 * sbn, 0, 1);
        }

#pragma unroll
        for (int t = 0; t < 4; ++t) {
            const float* Bp = Bk + (n0 + 16 * t + rc) * sbn;
            v16h b = fh_load_b_str(Bp, sbk);
            acc[0][t] = __builtin_amdgcn_wmma_f32_16x16x32_f16(
                false, a0, false, b, (short)0, acc[0][t], false, false);
            acc[1][t] = __builtin_amdgcn_wmma_f32_16x16x32_f16(
                false, a1, false, b, (short)0, acc[1][t], false, false);
        }
    }

#pragma unroll
    for (int t = 0; t < 4; ++t) {
        const long nn = n0 + 16 * t + rc;
        const float bv = bias ? bias[nn] : 0.f;
#pragma unroll
        for (int s = 0; s < 2; ++s) {
#pragma unroll
            for (int r = 0; r < 8; ++r) {
                long mm = m0 + 16 * s + r + 8 * grp;
                float v = acc[s][t][r] + bv;
                if (act) v = v > 0.f ? v : 0.01f * v;   // LeakyReLU(0.01)
                long idx = mm * ldd + nn;
                if (accum) v += D[idx];
                D[idx] = v;
            }
        }
    }
}

// ---------------------------------------------------------------------------
// Elementwise / reduction helpers
// ---------------------------------------------------------------------------
__global__ void fh_mvsum(const float* __restrict__ cm,
                         const float* __restrict__ mv,
                         float* __restrict__ mvsum)
{
    int c = blockIdx.x * blockDim.x + threadIdx.x;
    if (c >= CC) return;
    float s = 0.f;
    for (int i = 0; i < NN; ++i) s += cm[(long)i * CC + c] * mv[i];
    mvsum[c] = s;
}

__global__ void fh_make_s2c(const float* __restrict__ cm,
                            const float* __restrict__ mv,
                            const float* __restrict__ mvsum,
                            float* __restrict__ s2c)
{
    long idx = (long)blockIdx.x * blockDim.x + threadIdx.x;
    if (idx >= (long)NN * CC) return;
    int i = (int)(idx / CC), c = (int)(idx % CC);
    float m = cm[idx];
    s2c[idx] = (m * mv[i]) / (mvsum[c] * m + 1.f);
}

__global__ void fh_rowflag(const float* __restrict__ X, int rows, int cols,
                           float* __restrict__ flag)
{
    int r = blockIdx.x * blockDim.x + threadIdx.x;
    if (r >= rows) return;
    float s = 0.f;
    for (int j = 0; j < cols; ++j) s += X[(long)r * cols + j];
    flag[r] = (s != 0.f) ? 1.f : 0.f;
}

__global__ void fh_scale_rows(float* __restrict__ X, int cols,
                              const float* __restrict__ flag)
{
    int r = blockIdx.x;
    float f = flag[r];
    for (int j = threadIdx.x; j < cols; j += blockDim.x)
        X[(long)r * cols + j] *= f;
}

__global__ void fh_rownorm(const float* __restrict__ X, int rows, int cols,
                           float* __restrict__ nrm)
{
    int r = blockIdx.x * blockDim.x + threadIdx.x;
    if (r >= rows) return;
    float s = 0.f;
    for (int j = 0; j < cols; ++j) { float v = X[(long)r * cols + j]; s += v * v; }
    nrm[r] = sqrtf(s);
}

// softmax over axis 0 (down a column); one block per column
__global__ __launch_bounds__(256)
void fh_col_softmax(float* __restrict__ X, int rows, int cols)
{
    __shared__ float red[256];
    int c = blockIdx.x, t = threadIdx.x;
    float m = NEGBIG;
    for (int i = t; i < rows; i += 256) m = fmaxf(m, X[(long)i * cols + c]);
    red[t] = m; __syncthreads();
    for (int s = 128; s > 0; s >>= 1) { if (t < s) red[t] = fmaxf(red[t], red[t + s]); __syncthreads(); }
    m = red[0]; __syncthreads();
    float sum = 0.f;
    for (int i = t; i < rows; i += 256) {
        float e = __expf(X[(long)i * cols + c] - m);
        X[(long)i * cols + c] = e; sum += e;
    }
    red[t] = sum; __syncthreads();
    for (int s = 128; s > 0; s >>= 1) { if (t < s) red[t] += red[t + s]; __syncthreads(); }
    float inv = 1.f / red[0];
    for (int i = t; i < rows; i += 256) X[(long)i * cols + c] *= inv;
}

// cos-normalize + column mask + row softmax (in place); one block per row
__global__ __launch_bounds__(256)
void fh_cos_softmax_row(float* __restrict__ S, int cols,
                        const float* __restrict__ xn,
                        const float* __restrict__ yn,
                        const float* __restrict__ valid)
{
    __shared__ float red[256];
    int r = blockIdx.x, t = threadIdx.x;
    float xr = xn[r];
    long base = (long)r * cols;
    float m = NEGBIG;
    for (int j = t; j < cols; j += 256) {
        float den = xr * yn[j];
        if (den == 0.f) den = 1.f;                    // NaN(0/0) -> 0 rule
        float v = (valid[j] != 0.f) ? S[base + j] / den : NEGBIG;
        S[base + j] = v;
        m = fmaxf(m, v);
    }
    red[t] = m; __syncthreads();
    for (int s = 128; s > 0; s >>= 1) { if (t < s) red[t] = fmaxf(red[t], red[t + s]); __syncthreads(); }
    m = red[0]; __syncthreads();
    float sum = 0.f;
    for (int j = t; j < cols; j += 256) {
        float e = (valid[j] != 0.f) ? __expf(S[base + j] - m) : 0.f;
        S[base + j] = e; sum += e;
    }
    red[t] = sum; __syncthreads();
    for (int s = 128; s > 0; s >>= 1) { if (t < s) red[t] += red[t + s]; __syncthreads(); }
    float tot = red[0];
    float inv = (tot != 0.f) ? 1.f / tot : 0.f;
    for (int j = t; j < cols; j += 256) S[base + j] *= inv;
}

// cos-normalize the NxN h@h^T matrix, save diagonal, zero diagonal
__global__ void fh_cos_prep(float* __restrict__ S,
                            const float* __restrict__ hn,
                            float* __restrict__ diag)
{
    long total = (long)NN * NN;
    for (long idx = (long)blockIdx.x * blockDim.x + threadIdx.x; idx < total;
         idx += (long)gridDim.x * blockDim.x) {
        int i = (int)(idx >> 12);      // NN = 4096
        int j = (int)(idx & (NN - 1));
        float den = hn[i] * hn[j];
        if (den == 0.f) den = 1.f;
        float v = S[idx] / den;
        if (i == j) { diag[i] = v; v = 0.f; }
        S[idx] = v;
    }
}

// top-3 per row (strict >, first occurrence wins ties == lax.top_k)
__global__ void fh_topk3(const float* __restrict__ S,
                         int* __restrict__ tki, float* __restrict__ tkv)
{
    int i = blockIdx.x * blockDim.x + threadIdx.x;
    if (i >= NN) return;
    const float* row = S + (long)i * NN;
    float v0 = NEGBIG, v1 = NEGBIG, v2 = NEGBIG;
    int i0 = 0, i1 = 0, i2 = 0;
    for (int j = 0; j < NN; ++j) {
        float v = row[j];
        if (v > v0)      { v2 = v1; i2 = i1; v1 = v0; i1 = i0; v0 = v; i0 = j; }
        else if (v > v1) { v2 = v1; i2 = i1; v1 = v;  i1 = j; }
        else if (v > v2) { v2 = v;  i2 = j; }
    }
    tki[i * 3] = i0; tki[i * 3 + 1] = i1; tki[i * 3 + 2] = i2;
    tkv[i * 3] = v0; tkv[i * 3 + 1] = v1; tkv[i * 3 + 2] = v2;
}

// sparse hs2c.T @ h via atomics (3 nnz per row) + masked column sums
__global__ void fh_scatter(const int* __restrict__ tki,
                           const float* __restrict__ tkv,
                           const float* __restrict__ h,
                           float* __restrict__ hid2,
                           float* __restrict__ colsum)
{
    int i = blockIdx.x, n = threadIdx.x;     // 128 threads
    float hv = h[(long)i * HH + n];
#pragma unroll
    for (int t = 0; t < 3; ++t) {
        int   j = tki[i * 3 + t];
        float v = tkv[i * 3 + t];
        atomicAdd(&hid2[(long)j * HH + n], v * hv);
    }
    if (n < 3) atomicAdd(&colsum[tki[i * 3 + n]], tkv[i * 3 + n]);
}

// add diag(colsum!=0 ? diag : 0) contribution: hid2[j] += d*h[j]
__global__ void fh_add_diag(float* __restrict__ hid2,
                            const float* __restrict__ colsum,
                            const float* __restrict__ diag,
                            const float* __restrict__ h)
{
    int j = blockIdx.x, n = threadIdx.x;
    float d = (colsum[j] != 0.f) ? diag[j] : 0.f;
    hid2[(long)j * HH + n] += d * h[(long)j * HH + n];
}

__global__ void fh_sub(const float* __restrict__ a, const float* __restrict__ b,
                       float* __restrict__ o, long n)
{
    for (long i = (long)blockIdx.x * blockDim.x + threadIdx.x; i < n;
         i += (long)gridDim.x * blockDim.x)
        o[i] = a[i] - b[i];
}

__global__ void fh_final(const float* __restrict__ osum,
                         const float* __restrict__ Wout,
                         const float* __restrict__ bout,
                         float* __restrict__ out)
{
    int i = blockIdx.x * blockDim.x + threadIdx.x;
    if (i >= NN) return;
    float s = 0.f;
    for (int n = 0; n < HH; ++n) s += osum[(long)i * HH + n] * Wout[n];
    out[i] = s + bout[0];
}

// ---------------------------------------------------------------------------
extern "C" void kernel_launch(void* const* d_in, const int* in_sizes, int n_in,
                              void* d_out, int out_size, void* d_ws, size_t ws_size,
                              hipStream_t stream)
{
    (void)in_sizes; (void)n_in; (void)out_size; (void)ws_size;
    const float* x   = (const float*)d_in[0];   // [N,H]
    const float* cm  = (const float*)d_in[1];   // [N,C]
    const float* mv  = (const float*)d_in[2];   // [N]
    const float* W_ps      = (const float*)d_in[3];
    const float* b_ps      = (const float*)d_in[4];
    const float* W_ps_fore = (const float*)d_in[5];
    const float* b_ps_fore = (const float*)d_in[6];
    const float* W_ps_back = (const float*)d_in[7];
    const float* b_ps_back = (const float*)d_in[8];
    const float* W_hs      = (const float*)d_in[9];
    const float* b_hs      = (const float*)d_in[10];
    const float* W_hs_fore = (const float*)d_in[11];
    const float* b_hs_fore = (const float*)d_in[12];
    const float* W_hs_back = (const float*)d_in[13];
    const float* b_hs_back = (const float*)d_in[14];
    const float* W_indi    = (const float*)d_in[15];
    const float* b_indi    = (const float*)d_in[16];
    const float* W_out     = (const float*)d_in[17];
    const float* b_out     = (const float*)d_in[18];
    float* out = (float*)d_out;

    // ---- workspace carve (floats) ----
    float* ws = (float*)d_ws;
    size_t o = 0;
    float* hsim   = ws + o; o += (size_t)NN * NN;   // 64 MB: sim matrix (reused)
    float* bufNC  = ws + o; o += (size_t)NN * CC;   // s2c / logits / c2s
    float* hidC   = ws + o; o += (size_t)CC * HH;   // hidden (concept) [C,H]
    float* tmp1   = ws + o; o += (size_t)NN * HH;
    float* pshr   = ws + o; o += (size_t)NN * HH;
    float* pback  = ws + o; o += (size_t)NN * HH;
    float* hbuf   = ws + o; o += (size_t)NN * HH;   // h = x - p_back
    float* hid2   = ws + o; o += (size_t)NN * HH;   // hidden (stock) [N,H]
    float* hshr   = ws + o; o += (size_t)NN * HH;
    float* hback  = ws + o; o += (size_t)NN * HH;
    float* osum   = ws + o; o += (size_t)NN * HH;   // out_ps+out_hs+out_indi
    float* mvsum  = ws + o; o += CC;
    float* valid1 = ws + o; o += CC;
    float* ynormC = ws + o; o += CC;
    float* xnorm  = ws + o; o += NN;
    float* hnorm  = ws + o; o += NN;
    float* h2norm = ws + o; o += NN;
    float* valid2 = ws + o; o += NN;
    float* diag   = ws + o; o += NN;
    float* colsum = ws + o; o += NN;
    float* tkv    = ws + o; o += (size_t)NN * 3;
    int*   tki    = (int*)(ws + o); o += (size_t)NN * 3;

    auto gemm = [&](const float* A, long sam, long sak,
                    const float* B, long sbk, long sbn,
                    const float* bias, float* D, long ldd,
                    int M, int Ncol, int Ktot, int act, int accum) {
        dim3 grid(Ncol / 64, M / 32);
        if (sak == 1 && sbn == 1)
            fh_gemm_wmma<true, true><<<grid, 32, 0, stream>>>(A, sam, sak, B, sbk, sbn,
                bias, D, ldd, M, Ncol, Ktot, act, accum);
        else if (sak == 1)
            fh_gemm_wmma<true, false><<<grid, 32, 0, stream>>>(A, sam, sak, B, sbk, sbn,
                bias, D, ldd, M, Ncol, Ktot, act, accum);
        else if (sbn == 1)
            fh_gemm_wmma<false, true><<<grid, 32, 0, stream>>>(A, sam, sak, B, sbk, sbn,
                bias, D, ldd, M, Ncol, Ktot, act, accum);
        else
            fh_gemm_wmma<false, false><<<grid, 32, 0, stream>>>(A, sam, sak, B, sbk, sbn,
                bias, D, ldd, M, Ncol, Ktot, act, accum);
    };

    // ===== predefined-concept (ps) branch =====
    fh_mvsum<<<dim3((CC + 127) / 128), 128, 0, stream>>>(cm, mv, mvsum);
    fh_make_s2c<<<dim3(((long)NN * CC + 255) / 256), 256, 0, stream>>>(cm, mv, mvsum, bufNC);
    // hidden1 = s2c^T @ x              [C,H]
    gemm(bufNC, 1, CC, x, HH, 1, nullptr, hidC, HH, CC, HH, NN, 0, 0);
    fh_rowflag<<<dim3((CC + 127) / 128), 128, 0, stream>>>(hidC, CC, HH, valid1);
    // logits = x @ hidden1^T           [N,C]
    gemm(x, HH, 1, hidC, 1, HH, nullptr, bufNC, CC, NN, CC, HH, 0, 0);
    fh_col_softmax<<<dim3(CC), 256, 0, stream>>>(bufNC, NN, CC);            // s2c2
    // hidden2 = s2c2^T @ x, * valid1   [C,H]
    gemm(bufNC, 1, CC, x, HH, 1, nullptr, hidC, HH, CC, HH, NN, 0, 0);
    fh_scale_rows<<<dim3(CC), 128, 0, stream>>>(hidC, HH, valid1);
    fh_rownorm<<<dim3((NN + 127) / 128), 128, 0, stream>>>(x, NN, HH, xnorm);
    fh_rownorm<<<dim3((CC + 127) / 128), 128, 0, stream>>>(hidC, CC, HH, ynormC);
    // c2s = row_softmax(mask(cos_sim(x, hidden2)))
    gemm(x, HH, 1, hidC, 1, HH, nullptr, bufNC, CC, NN, CC, HH, 0, 0);
    fh_cos_softmax_row<<<dim3(NN), 256, 0, stream>>>(bufNC, CC, xnorm, ynormC, valid1);
    // p_shared = (c2s @ hidden2) @ W_ps^T + b_ps
    gemm(bufNC, CC, 1, hidC, HH, 1, nullptr, tmp1, HH, NN, HH, CC, 0, 0);
    gemm(tmp1, HH, 1, W_ps, 1, HH, b_ps, pshr, HH, NN, HH, HH, 0, 0);
    gemm(pshr, HH, 1, W_ps_back, 1, HH, b_ps_back, pback, HH, NN, HH, HH, 0, 0);
    gemm(pshr, HH, 1, W_ps_fore, 1, HH, b_ps_fore, osum, HH, NN, HH, HH, 1, 0); // out_ps

    // ===== hidden-concept (hs) branch =====
    fh_sub<<<dim3(2048), 256, 0, stream>>>(x, pback, hbuf, (long)NN * HH);  // h
    fh_rownorm<<<dim3((NN + 127) / 128), 128, 0, stream>>>(hbuf, NN, HH, hnorm);
    // hsim = h @ h^T                   [N,N]
    gemm(hbuf, HH, 1, hbuf, 1, HH, nullptr, hsim, NN, NN, NN, HH, 0, 0);
    fh_cos_prep<<<dim3(65536), 256, 0, stream>>>(hsim, hnorm, diag);
    hipMemsetAsync(hid2, 0, (size_t)NN * HH * sizeof(float), stream);
    hipMemsetAsync(colsum, 0, (size_t)NN * sizeof(float), stream);
    fh_topk3<<<dim3(NN / 128), 128, 0, stream>>>(hsim, tki, tkv);
    fh_scatter<<<dim3(NN), 128, 0, stream>>>(tki, tkv, hbuf, hid2, colsum);
    fh_add_diag<<<dim3(NN), HH, 0, stream>>>(hid2, colsum, diag, hbuf);
    fh_rowflag<<<dim3((NN + 127) / 128), 128, 0, stream>>>(hid2, NN, HH, valid2);
    fh_scale_rows<<<dim3(NN), 128, 0, stream>>>(hid2, HH, valid2);
    fh_rownorm<<<dim3((NN + 127) / 128), 128, 0, stream>>>(hid2, NN, HH, h2norm);
    // hc2s = row_softmax(mask(cos_sim(h, hidden2)))   (reuse hsim buffer)
    gemm(hbuf, HH, 1, hid2, 1, HH, nullptr, hsim, NN, NN, NN, HH, 0, 0);
    fh_cos_softmax_row<<<dim3(NN), 256, 0, stream>>>(hsim, NN, hnorm, h2norm, valid2);
    // h_shared = (hc2s @ hidden2) @ W_hs^T + b_hs
    gemm(hsim, NN, 1, hid2, HH, 1, nullptr, tmp1, HH, NN, HH, NN, 0, 0);
    gemm(tmp1, HH, 1, W_hs, 1, HH, b_hs, hshr, HH, NN, HH, HH, 0, 0);
    gemm(hshr, HH, 1, W_hs_back, 1, HH, b_hs_back, hback, HH, NN, HH, HH, 0, 0);
    gemm(hshr, HH, 1, W_hs_fore, 1, HH, b_hs_fore, osum, HH, NN, HH, HH, 1, 1);  // += out_hs

    // ===== individual branch + head =====
    fh_sub<<<dim3(2048), 256, 0, stream>>>(hbuf, hback, tmp1, (long)NN * HH); // indiv
    gemm(tmp1, HH, 1, W_indi, 1, HH, b_indi, osum, HH, NN, HH, HH, 1, 1);     // += out_indi
    fh_final<<<dim3(NN / 128), 128, 0, stream>>>(osum, W_out, b_out, out);
}
